// LSTM2d_41540923687175
// MI455X (gfx1250) — matrix operand order repaired and checked by reading, not compiled, not run on hardware
//
#include <hip/hip_runtime.h>
#include <hip/hip_bf16.h>
#include <math.h>

typedef __attribute__((ext_vector_type(16))) _Float16 v16h;
typedef __attribute__((ext_vector_type(8)))  float    v8f;

#define LSEQ   64
#define NB     32
#define EMB    128
#define VOCAB  32000

// ---------------------------------------------------------------------------
// WMMA helpers (CDNA5 V_WMMA_F32_16X16X32_F16)
// ---------------------------------------------------------------------------
__device__ __forceinline__ v8f wmma_f16(v16h a, v16h b, v8f c) {
  return __builtin_amdgcn_wmma_f32_16x16x32_f16(false, a, false, b, (short)0, c,
                                                false, false);
}

// A-fragment (16x32 MxK, fp16) element position for (row, k), tiles ordered
// [m][kt].  ISA layout: lanes 0-15 hold K={0..7,16..23}; lanes 16-31 hold
// K={8..15,24..31}; within a lane halves are K-minor.
__device__ __forceinline__ int afrag_idx(int row, int k, int ktiles) {
  const int kl = k & 31;
  return ((((row >> 4) * ktiles) + (k >> 5)) << 9) +
         (((row & 15) + (((kl >> 3) & 1) << 4)) << 4) +
         (kl & 7) + ((kl >> 4) << 3);
}

// Gather a 16x32 fp16 A-fragment from an f32 row-major source (cold paths).
__device__ __forceinline__ v16h gather_a(const float* __restrict__ src, int ld,
                                         int mbase, int kbase, int lane) {
  const float* p = src + (size_t)(mbase + (lane & 15)) * ld + kbase +
                   ((lane & 16) ? 8 : 0);
  v16h a;
#pragma unroll
  for (int t = 0; t < 8; ++t) a[t] = (_Float16)p[t];
#pragma unroll
  for (int t = 0; t < 8; ++t) a[t + 8] = (_Float16)p[16 + t];
  return a;
}

// Pre-packed 32x16 (KxN) fp16 B tile: one 32-byte v16h load per lane.
__device__ __forceinline__ v16h load_b(const _Float16* __restrict__ bt, int lane) {
  return *(const v16h*)(bt + lane * 16);
}

__device__ __forceinline__ float sigmf(float x) { return 1.0f / (1.0f + expf(-x)); }

// Wave-relative LDS byte offset of a __shared__ pointer (generic addr low 32b).
__device__ __forceinline__ unsigned lds_off32(const void* p) {
  return (unsigned)(unsigned long long)(uintptr_t)p;
}

// Async global->LDS copy, 16B per lane per issue (ASYNCcnt-tracked).
__device__ __forceinline__ void async_copy_b128(unsigned ldsOff, const void* g) {
  asm volatile("global_load_async_to_lds_b128 %0, %1, off"
               :: "v"(ldsOff), "v"((unsigned long long)(uintptr_t)g)
               : "memory");
}

// ---------------------------------------------------------------------------
// Weight packing: W[N,K] (row-major f32, used as x @ W^T) -> fp16 B tiles.
// ---------------------------------------------------------------------------
__global__ __launch_bounds__(256) void pack_b_kernel(
    const float* __restrict__ W, int srcld, int colOff, int K, long total,
    _Float16* __restrict__ dst) {
  long e = (long)blockIdx.x * 256 + threadIdx.x;
  if (e >= total) return;
  int ktiles = K >> 5;
  long tile = e >> 9;
  int within = (int)(e & 511);
  int lane = within >> 4, half = within & 15;
  long nt = tile / ktiles;
  int  kt = (int)(tile % ktiles);
  long n = nt * 16 + (lane & 15);
  int  k = kt * 32 + ((lane >> 4) << 3) + (half & 7) + ((half >> 3) << 4);
  dst[e] = (_Float16)W[n * srcld + colOff + k];
}

__global__ void bias_combine_kernel(const float* bihf, const float* bhhf,
                                    const float* bihb, const float* bhhb,
                                    const float* bx, const float* bs,
                                    float* bF, float* bB, float* bXS) {
  int t = threadIdx.x;  // 640
  if (t < 512) { bF[t] = bihf[t] + bhhf[t]; bB[t] = bihb[t] + bhhb[t]; }
  bXS[t] = bx[t] + bs[t];
}

__global__ __launch_bounds__(256) void zero_kernel(float* p, long n) {
  long i = (long)blockIdx.x * 256 + threadIdx.x;
  if (i < n) p[i] = 0.f;
}

__global__ __launch_bounds__(256) void embed_kernel(const int* __restrict__ x,
                                                    const float* __restrict__ em,
                                                    float* __restrict__ out) {
  long idx = (long)blockIdx.x * 256 + threadIdx.x;  // 64*32*128
  if (idx >= (long)LSEQ * NB * EMB) return;
  long tb = idx >> 7;
  int  e  = (int)(idx & 127);
  out[idx] = em[(size_t)x[tb] * EMB + e];
}

// ---------------------------------------------------------------------------
// Generic GEMM (cold, one-time):  C[rb][32,N] = A[rb][32,K](f32) * Bt + bias
// ---------------------------------------------------------------------------
__global__ __launch_bounds__(256) void gemm_rows32(
    const float* __restrict__ A, int lda, long aBS,
    const _Float16* __restrict__ Bt, int ktiles, int ntilesTotal, int tps,
    const float* __restrict__ bias, float* __restrict__ C, int ldc, long cBS,
    int nslabs) {
  const int rb = blockIdx.x / nslabs;
  const int slab = blockIdx.x % nslabs;
  const float* Ab = A + (size_t)rb * aBS;
  float* Cb = C + (size_t)rb * cBS;
  const int lane = threadIdx.x & 31;
  const int wave = threadIdx.x >> 5;
  const int nwaves = blockDim.x >> 5;
  const int nt0 = slab * tps;
  int nt1 = nt0 + tps; if (nt1 > ntilesTotal) nt1 = ntilesTotal;
  for (int nt = nt0 + wave; nt < nt1; nt += nwaves) {
    const _Float16* bt = Bt + (size_t)nt * ktiles * 512;
    for (int m = 0; m < 2; ++m) {
      v8f acc = {};
      for (int kt = 0; kt < ktiles; ++kt) {
        v16h a = gather_a(Ab, lda, m * 16, kt * 32, lane);
        v16h b = load_b(bt + (size_t)kt * 512, lane);
        acc = wmma_f16(a, b, acc);
      }
      const int col = nt * 16 + (lane & 15);
      const int roff = (lane & 16) ? 8 : 0;
      const float bv = bias ? bias[col] : 0.f;
#pragma unroll
      for (int r = 0; r < 8; ++r)
        Cb[(size_t)(m * 16 + roff + r) * ldc + col] = acc[r] + bv;
    }
  }
}

// ---------------------------------------------------------------------------
// Logits GEMM: A already in fp16 fragment layout (K=128 -> 4 k-tiles).
// ---------------------------------------------------------------------------
__global__ __launch_bounds__(256) void gemm_logits_af16(
    const _Float16* __restrict__ Af, const _Float16* __restrict__ Bt,
    const float* __restrict__ bias, float* __restrict__ C) {
  const int lane = threadIdx.x & 31;
  const int wave = threadIdx.x >> 5;
  const int nt0 = blockIdx.x * 16;
  for (int nt = nt0 + wave; nt < nt0 + 16; nt += 8) {
    const _Float16* bt = Bt + (size_t)nt * 4 * 512;
    v16h bf[4];
#pragma unroll
    for (int kt = 0; kt < 4; ++kt) bf[kt] = load_b(bt + (size_t)kt * 512, lane);
#pragma unroll
    for (int m = 0; m < 2; ++m) {
      v8f acc = {};
#pragma unroll
      for (int kt = 0; kt < 4; ++kt)
        acc = wmma_f16(*(const v16h*)(Af + ((m * 4 + kt) << 9) + lane * 16),
                       bf[kt], acc);
      const int col = nt * 16 + (lane & 15);
      const int roff = (lane & 16) ? 8 : 0;
      const float bv = bias[col];
#pragma unroll
      for (int r = 0; r < 8; ++r)
        C[(size_t)(m * 16 + roff + r) * VOCAB + col] = acc[r] + bv;
    }
  }
}

// Per-row: ypart = y_i @ WxY^T  (fragment-layout A, 1 block, 8 waves).
__global__ __launch_bounds__(256) void ypart_kernel(
    const _Float16* __restrict__ ybufF, const _Float16* __restrict__ WxYp,
    float* __restrict__ ypg) {
  const int lane = threadIdx.x & 31;
  const int wave = threadIdx.x >> 5;
  for (int nt = wave; nt < 40; nt += 8) {
    const _Float16* bt = WxYp + (size_t)nt * 4 * 512;
    v16h bf[4];
#pragma unroll
    for (int kt = 0; kt < 4; ++kt) bf[kt] = load_b(bt + (size_t)kt * 512, lane);
#pragma unroll
    for (int m = 0; m < 2; ++m) {
      v8f acc = {};
#pragma unroll
      for (int kt = 0; kt < 4; ++kt)
        acc = wmma_f16(*(const v16h*)(ybufF + ((m * 4 + kt) << 9) + lane * 16),
                       bf[kt], acc);
      const int col = nt * 16 + (lane & 15);
      const int roff = (lane & 16) ? 8 : 0;
#pragma unroll
      for (int r = 0; r < 8; ++r) ypg[(m * 16 + roff + r) * 640 + col] = acc[r];
    }
  }
}

// Per-row: xprow[j] = xpart[j] + ypart (broadcast over j).
__global__ __launch_bounds__(256) void xprow_kernel(
    const float* __restrict__ xpart, const float* __restrict__ ypg,
    float* __restrict__ xprow) {
  long e = (long)blockIdx.x * 256 + threadIdx.x;  // 64*32*640
  if (e >= (long)LSEQ * NB * 640) return;
  int rc = (int)(e % (NB * 640));
  xprow[e] = xpart[e] + ypg[rc];
}

// ---------------------------------------------------------------------------
// Persistent bidirectional encoder (fp16 fragment hidden state in LDS).
// ---------------------------------------------------------------------------
__global__ __launch_bounds__(256) void encoder_kernel(
    const float* __restrict__ Xf, const float* __restrict__ Xb,
    const _Float16* __restrict__ WhhF, const _Float16* __restrict__ WhhB,
    float* __restrict__ hcat) {
  __shared__ float z[32 * 512];
  __shared__ __align__(32) _Float16 hF[2 * 4 * 512];
  __shared__ float c[32 * 128];
  const int dir = blockIdx.x;
  const float* X = dir ? Xb : Xf;
  const _Float16* Whh = dir ? WhhB : WhhF;
  const int tid = threadIdx.x, lane = tid & 31, wave = tid >> 5;
  for (int i = tid; i < 32 * 128; i += 256) c[i] = 0.f;
  for (int i = tid; i < 4096; i += 256) hF[i] = (_Float16)0.f;
  __syncthreads();
  for (int step = 0; step < 64; ++step) {
    const int t = dir ? (63 - step) : step;
    const float* Xt = X + (size_t)t * 32 * 512;
    for (int nt = wave; nt < 32; nt += 8) {
      const _Float16* bt = Whh + (size_t)nt * 4 * 512;
      v16h bf[4];
#pragma unroll
      for (int kt = 0; kt < 4; ++kt) bf[kt] = load_b(bt + (size_t)kt * 512, lane);
#pragma unroll
      for (int m = 0; m < 2; ++m) {
        v8f acc = {};
#pragma unroll
        for (int kt = 0; kt < 4; ++kt)
          acc = wmma_f16(*(const v16h*)(hF + ((m * 4 + kt) << 9) + lane * 16),
                         bf[kt], acc);
        const int col = nt * 16 + (lane & 15);
        const int roff = (lane & 16) ? 8 : 0;
#pragma unroll
        for (int r = 0; r < 8; ++r) {
          const int row = m * 16 + roff + r;
          z[row * 512 + col] = acc[r] + Xt[(size_t)row * 512 + col];
        }
      }
    }
    __syncthreads();
    for (int i = tid; i < 32 * 128; i += 256) {
      const int bb = i >> 7, jj = i & 127;
      const float* zr = z + bb * 512;
      const float cv = sigmf(zr[128 + jj]) * c[i] +
                       sigmf(zr[jj]) * tanhf(zr[256 + jj]);
      const float hv = sigmf(zr[384 + jj]) * tanhf(cv);
      c[i] = cv;
      hF[afrag_idx(bb, jj, 4)] = (_Float16)hv;
      hcat[(size_t)t * 32 * 256 + bb * 256 + dir * 128 + jj] = hv;
    }
    __syncthreads();
  }
}

// ---------------------------------------------------------------------------
// One decoder row: 64 sequential 2D-LSTM cells, 1 WG, 20 waves.
// xprow slabs are double-buffered into LDS via async global->LDS copies
// overlapped with the WMMA state GEMM of the current cell.
// ---------------------------------------------------------------------------
__global__ __launch_bounds__(640) void row_kernel(
    const float* __restrict__ xprow, const _Float16* __restrict__ WsPp,
    _Float16* __restrict__ s_rowF, float* __restrict__ c_row,
    _Float16* __restrict__ sLastF) {
  __shared__ float z[32 * 640];
  __shared__ __align__(16) float xbuf[2][32 * 640];
  __shared__ __align__(32) _Float16 shF[2 * 4 * 512];
  __shared__ float ch[32 * 128];
  const int tid = threadIdx.x, lane = tid & 31, wave = tid >> 5;  // 20 waves

  for (int i = tid; i < 32 * 128; i += 640) ch[i] = 0.f;
  for (int i = tid; i < 4096; i += 640) shF[i] = (_Float16)0.f;

  // Preload slab 0 (each thread: 8 x 16B, wave-uniform -> 8 asyncs/wave).
  {
    const unsigned dst = lds_off32(&xbuf[0][0]) + tid * 16;
    const char* src = (const char*)xprow + tid * 16;
#pragma unroll
    for (int s = 0; s < 8; ++s)
      async_copy_b128(dst + s * 10240, src + s * 10240);
  }
  __syncthreads();

  for (int j = 0; j < 64; ++j) {
    const int cur = j & 1;
    const int jn = (j + 1 < 64) ? (j + 1) : 63;  // uniform issue count
    {
      const unsigned dst = lds_off32(&xbuf[cur ^ 1][0]) + tid * 16;
      const char* src = (const char*)(xprow + (size_t)jn * 20480) + tid * 16;
#pragma unroll
      for (int s = 0; s < 8; ++s)
        async_copy_b128(dst + s * 10240, src + s * 10240);
    }
    // Retire previous batch (8 just-issued may remain in flight).
    asm volatile("s_wait_asynccnt 0x8" ::: "memory");
    __syncthreads();

    const _Float16* svF = s_rowF + (size_t)j * 4096;
    const float* xc = &xbuf[cur][0];
    for (int nt = wave; nt < 40; nt += 20) {
      const _Float16* bt = WsPp + (size_t)nt * 8 * 512;
      v16h bf[8];
#pragma unroll
      for (int kt = 0; kt < 8; ++kt) bf[kt] = load_b(bt + (size_t)kt * 512, lane);
#pragma unroll
      for (int m = 0; m < 2; ++m) {
        v8f acc = {};
#pragma unroll
        for (int kt = 0; kt < 4; ++kt)  // K 0..127 : s_hor (LDS)
          acc = wmma_f16(*(const v16h*)(shF + ((m * 4 + kt) << 9) + lane * 16),
                         bf[kt], acc);
#pragma unroll
        for (int kt = 0; kt < 4; ++kt)  // K 128..255 : s_ver (L2)
          acc = wmma_f16(*(const v16h*)(svF + ((m * 4 + kt) << 9) + lane * 16),
                         bf[4 + kt], acc);
        const int col = nt * 16 + (lane & 15);
        const int roff = (lane & 16) ? 8 : 0;
#pragma unroll
        for (int r = 0; r < 8; ++r) {
          const int row = m * 16 + roff + r;
          z[row * 640 + col] = acc[r] + xc[row * 640 + col];
        }
      }
    }
    __syncthreads();
    for (int i = tid; i < 32 * 128; i += 640) {
      const int bb = i >> 7, jj = i & 127;
      const float* zr = z + bb * 640;
      const float gi = sigmf(zr[jj]);
      const float gf = sigmf(zr[128 + jj]);
      const float go = sigmf(zr[256 + jj]);
      const float gl = sigmf(zr[384 + jj]);
      const float gg = tanhf(zr[512 + jj]);
      const float cver = c_row[(size_t)j * 4096 + i];
      const float cnew = gf * (gl * ch[i] + (1.f - gl) * cver) + gi * gg;
      const float snew = go * tanhf(cnew);
      ch[i] = cnew;
      const int fi = afrag_idx(bb, jj, 4);
      const _Float16 sh16 = (_Float16)snew;
      shF[fi] = sh16;                        // horizontal state for j+1
      s_rowF[(size_t)j * 4096 + fi] = sh16;  // vertical state for next row
      c_row[(size_t)j * 4096 + i] = cnew;
      if (j == 63) sLastF[fi] = sh16;        // feeds logits GEMM directly
    }
    __syncthreads();
  }
}

// ---------------------------------------------------------------------------
// Softmax reduction per batch row; also gathers y_{i+1} = embedding[argmax]
// directly into fp16 fragment layout (one block owns one batch row).
// ---------------------------------------------------------------------------
__global__ __launch_bounds__(256) void softmax_reduce(
    const float* __restrict__ zlog, const float* __restrict__ em,
    float* __restrict__ rowmax, float* __restrict__ rowrcp,
    _Float16* __restrict__ ybufF) {
  __shared__ float smax[256];
  __shared__ int   samax[256];
  __shared__ float ssum[256];
  const int bb = blockIdx.x, tid = threadIdx.x;
  const float* zr = zlog + (size_t)bb * VOCAB;
  float m = -3.4e38f; int am = 0;
  for (int v = tid; v < VOCAB; v += 256) {
    float xv = zr[v];
    if (xv > m) { m = xv; am = v; }
  }
  smax[tid] = m; samax[tid] = am;
  __syncthreads();
  for (int s = 128; s > 0; s >>= 1) {
    if (tid < s && smax[tid + s] > smax[tid]) {
      smax[tid] = smax[tid + s]; samax[tid] = samax[tid + s];
    }
    __syncthreads();
  }
  const float gm = smax[0];
  float sum = 0.f;
  for (int v = tid; v < VOCAB; v += 256) sum += expf(zr[v] - gm);
  ssum[tid] = sum;
  __syncthreads();
  for (int s = 128; s > 0; s >>= 1) {
    if (tid < s) ssum[tid] += ssum[tid + s];
    __syncthreads();
  }
  if (tid == 0) { rowmax[bb] = gm; rowrcp[bb] = 1.0f / ssum[0]; }
  const int amax = samax[0];
  if (tid < EMB)
    ybufF[afrag_idx(bb, tid, 4)] = (_Float16)em[(size_t)amax * EMB + tid];
}

__global__ __launch_bounds__(256) void softmax_write(
    const float* __restrict__ zlog, const float* __restrict__ rowmax,
    const float* __restrict__ rowrcp, float* __restrict__ out) {
  long idx = (long)blockIdx.x * 256 + threadIdx.x;  // 32*32000
  if (idx >= (long)NB * VOCAB) return;
  int bb = (int)(idx / VOCAB);
  out[idx] = expf(zlog[idx] - rowmax[bb]) * rowrcp[bb];
}

// ---------------------------------------------------------------------------
extern "C" void kernel_launch(void* const* d_in, const int* in_sizes, int n_in,
                              void* d_out, int out_size, void* d_ws,
                              size_t ws_size, hipStream_t stream) {
  const int*   x      = (const int*)d_in[0];
  const float* em     = (const float*)d_in[1];
  const float* Wih_f  = (const float*)d_in[2];
  const float* Whh_f  = (const float*)d_in[3];
  const float* bih_f  = (const float*)d_in[4];
  const float* bhh_f  = (const float*)d_in[5];
  const float* Wih_b  = (const float*)d_in[6];
  const float* Whh_b  = (const float*)d_in[7];
  const float* bih_b  = (const float*)d_in[8];
  const float* bhh_b  = (const float*)d_in[9];
  const float* Wx     = (const float*)d_in[10];
  const float* bx     = (const float*)d_in[11];
  const float* Ws     = (const float*)d_in[12];
  const float* bs     = (const float*)d_in[13];
  const float* logW   = (const float*)d_in[14];
  const float* logb   = (const float*)d_in[15];
  float* out = (float*)d_out;

  char* wsb = (char*)d_ws;
  size_t cur = 0;
  auto alloc = [&](size_t bytes) -> void* {
    void* p = wsb + cur;
    cur = (cur + bytes + 255) & ~(size_t)255;
    return p;
  };

  _Float16* WfIHp = (_Float16*)alloc((size_t)512 * 128 * 2);
  _Float16* WfHHp = (_Float16*)alloc((size_t)512 * 128 * 2);
  _Float16* WbIHp = (_Float16*)alloc((size_t)512 * 128 * 2);
  _Float16* WbHHp = (_Float16*)alloc((size_t)512 * 128 * 2);
  _Float16* WxYp  = (_Float16*)alloc((size_t)640 * 128 * 2);
  _Float16* WxHp  = (_Float16*)alloc((size_t)640 * 256 * 2);
  _Float16* WsPp  = (_Float16*)alloc((size_t)640 * 256 * 2);
  _Float16* WlogP = (_Float16*)alloc((size_t)VOCAB * 128 * 2);
  float* bF    = (float*)alloc(512 * 4);
  float* bB    = (float*)alloc(512 * 4);
  float* bXS   = (float*)alloc(640 * 4);
  float* emb   = (float*)alloc((size_t)LSEQ * NB * EMB * 4);
  float* Xf    = (float*)alloc((size_t)LSEQ * NB * 512 * 4);
  float* Xb    = (float*)alloc((size_t)LSEQ * NB * 512 * 4);
  float* hcat  = (float*)alloc((size_t)LSEQ * NB * 256 * 4);
  float* xpart = (float*)alloc((size_t)LSEQ * NB * 640 * 4);
  float* xprow = (float*)alloc((size_t)LSEQ * NB * 640 * 4);
  float* ypg   = (float*)alloc((size_t)NB * 640 * 4);
  _Float16* s_rowF = (_Float16*)alloc((size_t)LSEQ * 4096 * 2);
  float*    c_row  = (float*)alloc((size_t)LSEQ * NB * 128 * 4);
  _Float16* ybufF  = (_Float16*)alloc((size_t)4096 * 2);
  _Float16* sLastF = (_Float16*)alloc((size_t)4096 * 2);
  float* zlog  = (float*)alloc((size_t)NB * VOCAB * 4);
  float* rmax  = (float*)alloc(32 * 4);
  float* rrcp  = (float*)alloc(32 * 4);
  if (cur > ws_size) return;  // workspace too small: bail deterministically

  auto gr = [](long n) { return (unsigned)((n + 255) / 256); };

  // ---- weight packing (fp16 B tiles) ----
  pack_b_kernel<<<gr(512L * 128), 256, 0, stream>>>(Wih_f, 128, 0, 128, 512L * 128, WfIHp);
  pack_b_kernel<<<gr(512L * 128), 256, 0, stream>>>(Whh_f, 128, 0, 128, 512L * 128, WfHHp);
  pack_b_kernel<<<gr(512L * 128), 256, 0, stream>>>(Wih_b, 128, 0, 128, 512L * 128, WbIHp);
  pack_b_kernel<<<gr(512L * 128), 256, 0, stream>>>(Whh_b, 128, 0, 128, 512L * 128, WbHHp);
  pack_b_kernel<<<gr(640L * 128), 256, 0, stream>>>(Wx, 384, 0, 128, 640L * 128, WxYp);
  pack_b_kernel<<<gr(640L * 256), 256, 0, stream>>>(Wx, 384, 128, 256, 640L * 256, WxHp);
  pack_b_kernel<<<gr(640L * 256), 256, 0, stream>>>(Ws, 256, 0, 256, 640L * 256, WsPp);
  pack_b_kernel<<<gr((long)VOCAB * 128), 256, 0, stream>>>(logW, 128, 0, 128, (long)VOCAB * 128, WlogP);
  bias_combine_kernel<<<1, 640, 0, stream>>>(bih_f, bhh_f, bih_b, bhh_b, bx, bs, bF, bB, bXS);

  // ---- init state (fp16 buffers zeroed as dword writes) ----
  zero_kernel<<<gr(2048), 256, 0, stream>>>((float*)ybufF, 2048);
  zero_kernel<<<gr((long)LSEQ * 2048), 256, 0, stream>>>((float*)s_rowF, (long)LSEQ * 2048);
  zero_kernel<<<gr((long)LSEQ * NB * 128), 256, 0, stream>>>(c_row, (long)LSEQ * NB * 128);

  // ---- embedding + encoder input projections (big parallel GEMMs) ----
  embed_kernel<<<gr((long)LSEQ * NB * EMB), 256, 0, stream>>>(x, em, emb);
  gemm_rows32<<<64, 256, 0, stream>>>(emb, 128, 32L * 128, WfIHp, 4, 32, 32, bF,
                                      Xf, 512, 32L * 512, 1);
  gemm_rows32<<<64, 256, 0, stream>>>(emb, 128, 32L * 128, WbIHp, 4, 32, 32, bB,
                                      Xb, 512, 32L * 512, 1);

  // ---- bidirectional encoder recurrence ----
  encoder_kernel<<<2, 256, 0, stream>>>(Xf, Xb, WfHHp, WbHHp, hcat);

  // ---- decoder precompute: xpart[j] = h_j @ WxH^T + bx + bs ----
  gemm_rows32<<<64, 256, 0, stream>>>(hcat, 256, 32L * 256, WxHp, 8, 40, 40, bXS,
                                      xpart, 640, 32L * 640, 1);

  // ---- 64 output rows (strictly sequential: y_i depends on row i-1 argmax) --
  for (int i = 0; i < LSEQ; ++i) {
    ypart_kernel<<<1, 256, 0, stream>>>(ybufF, WxYp, ypg);
    xprow_kernel<<<gr((long)LSEQ * NB * 640), 256, 0, stream>>>(xpart, ypg, xprow);
    row_kernel<<<1, 640, 0, stream>>>(xprow, WsPp, s_rowF, c_row, sLastF);
    gemm_logits_af16<<<125, 256, 0, stream>>>(sLastF, WlogP, logb, zlog);
    softmax_reduce<<<32, 256, 0, stream>>>(zlog, em, rmax, rrcp, ybufF);
    softmax_write<<<gr((long)NB * VOCAB), 256, 0, stream>>>(
        zlog, rmax, rrcp, out + (size_t)i * NB * VOCAB);
  }
}
